// DynamicConv_2997887172927
// MI455X (gfx1250) — matrix operand order repaired and checked by reading, not compiled, not run on hardware
//
#include <hip/hip_runtime.h>

#define N_    32
#define CI_   256
#define CO_   256
#define KS_   3
#define H_    64
#define W_    64
#define M_    4
#define HID_  64
#define KTOT_ (CI_*KS_*KS_)   /* 2304 = 9 taps * 256 ci */
#define TAU_  (1.0f/30.0f)

typedef _Float16 h16;
typedef __attribute__((ext_vector_type(16))) _Float16     v16h;
typedef __attribute__((ext_vector_type(8)))  float        v8f;
typedef __attribute__((ext_vector_type(4)))  unsigned int v4u;
typedef int v4i_ __attribute__((vector_size(16)));        // matches builtin prototype

// ---- CDNA5 async global->LDS copy (GLOBAL_LOAD_ASYNC_TO_LDS_B128, ASYNCcnt) ----
#if __has_builtin(__builtin_amdgcn_global_load_async_to_lds_b128)
  #define GLB_V4I(p)                                                       \
      ((__attribute__((address_space(1))) v4i_*)                           \
       (__attribute__((address_space(1))) void*)(const void*)(p))
  #define LDS_V4I(p)                                                       \
      ((__attribute__((address_space(3))) v4i_*)                           \
       (unsigned int)(unsigned long long)(p))
  #define ASYNC_CP16(g, l, OFF)                                            \
      __builtin_amdgcn_global_load_async_to_lds_b128(GLB_V4I(g), LDS_V4I(l), (OFF), 0)
  #if __has_builtin(__builtin_amdgcn_s_wait_asynccnt)
    #define WAIT_ASYNC() __builtin_amdgcn_s_wait_asynccnt(0)
  #else
    #define WAIT_ASYNC() asm volatile("s_wait_asynccnt 0" ::: "memory")
  #endif
#else
  #define ASYNC_CP16(g, l, OFF)                                            \
      (*reinterpret_cast<v4u*>((char*)(l) + (OFF)) =                       \
       *reinterpret_cast<const v4u*>((const char*)(g) + (OFF)))
  #define WAIT_ASYNC() ((void)0)
#endif

__device__ __forceinline__ int tap_kh(int tap) { return tap < 3 ? 0 : (tap < 6 ? 1 : 2); }

// ---------------- Kernel A: global average pool: pooled[n,ci] ----------------
__global__ __launch_bounds__(256) void k_pool(const float* __restrict__ x,
                                              float* __restrict__ pooled) {
    __shared__ float red[256];
    const int b = blockIdx.x;                 // n*CI + ci
    const float4* p4 = reinterpret_cast<const float4*>(x + (size_t)b * (H_*W_));
    float s = 0.f;
    for (int i = threadIdx.x; i < (H_*W_)/4; i += 256) {
        float4 v = p4[i];
        s += (v.x + v.y) + (v.z + v.w);
    }
    red[threadIdx.x] = s;
    __syncthreads();
    for (int off = 128; off > 0; off >>= 1) {
        if ((int)threadIdx.x < off) red[threadIdx.x] += red[threadIdx.x + off];
        __syncthreads();
    }
    if (threadIdx.x == 0) pooled[b] = red[0] * (1.0f/(H_*W_));
}

// ------------- Kernel B: attention MLP + softmax + bias aggregation ----------
__global__ __launch_bounds__(64) void k_attn(const float* __restrict__ pooled,
                                             const float* __restrict__ w1,
                                             const float* __restrict__ b1,
                                             const float* __restrict__ w2,
                                             const float* __restrict__ b2,
                                             const float* __restrict__ Bbank,
                                             float* __restrict__ pi,
                                             float* __restrict__ bn) {
    __shared__ float hmid[HID_];
    __shared__ float pish[M_];
    const int n = blockIdx.x;
    const int t = threadIdx.x;                // 64 threads
    {
        float s = b1[t];
        const float* pv = pooled + n*CI_;
        const float* wr = w1 + t*CI_;
        for (int c = 0; c < CI_; ++c) s += pv[c]*wr[c];
        hmid[t] = fmaxf(s, 0.f);
    }
    __syncthreads();
    if (t < M_) {
        float lg = b2[t];
        const float* w2r = w2 + t*HID_;
        for (int j = 0; j < HID_; ++j) lg += hmid[j]*w2r[j];
        pish[t] = lg * TAU_;
    }
    __syncthreads();
    if (t == 0) {
        float mx = pish[0];
        for (int m = 1; m < M_; ++m) mx = fmaxf(mx, pish[m]);
        float e[M_], sum = 0.f;
        for (int m = 0; m < M_; ++m) { e[m] = __expf(pish[m]-mx); sum += e[m]; }
        float inv = 1.0f/sum;
        for (int m = 0; m < M_; ++m) { pish[m] = e[m]*inv; pi[n*M_+m] = pish[m]; }
    }
    __syncthreads();
    for (int co = t; co < CO_; co += HID_) {
        float bb = 0.f;
        for (int m = 0; m < M_; ++m) bb += pish[m]*Bbank[co*M_+m];
        bn[n*CO_+co] = bb;
    }
}

// --------- Kernel C: aggregate per-sample weights -> f16 ---------------------
// Output K-order is (tap, ci): wagg[n][co][tap*256 + ci], tap = kh*3+kw.
// Coalesced global reads -> LDS stage -> conflict-free gather (stride 9 vs 64 banks).
__global__ __launch_bounds__(256) void k_aggw(const float* __restrict__ Wbank,
                                              const float* __restrict__ pi,
                                              h16* __restrict__ wagg) {
    __shared__ float sw[M_][KTOT_];           // 36864 B
    const int b  = blockIdx.x;                // n*CO + co
    const int n  = b / CO_;
    const int co = b - n*CO_;
    const int t  = threadIdx.x;
    const float p0 = pi[n*M_+0], p1 = pi[n*M_+1], p2 = pi[n*M_+2], p3 = pi[n*M_+3];
    const float* src = Wbank + (size_t)co*M_*KTOT_;
    #pragma unroll
    for (int m = 0; m < M_; ++m)
        for (int i = t; i < KTOT_; i += 256) sw[m][i] = src[m*KTOT_ + i];
    __syncthreads();
    h16* dst = wagg + (size_t)b*KTOT_;
    for (int i = t; i < KTOT_; i += 256) {
        const int tap = i >> 8;               // 0..8
        const int ci  = i & 255;
        const int s   = ci*9 + tap;           // source (ci,kh,kw) order
        float v = p0*sw[0][s] + p1*sw[1][s] + p2*sw[2][s] + p3*sw[3][s];
        dst[i] = (h16)v;
    }
}

// -------- Kernel D: implicit-GEMM conv via v_wmma_f32_16x16x32_f16 -----------
// Block = 256 threads (8 waves, 4x2): tile 128 co x 64 pos, K staged 64/round.
// Wave: 32co x 32pos -> 8 wmma per round. A-tile: async global->LDS (dbl buf).
__global__ __launch_bounds__(256) void k_conv(const float* __restrict__ x,
                                              const h16*  __restrict__ wagg,
                                              const float* __restrict__ bn,
                                              float* __restrict__ y) {
    constexpr int AST = 72;                   // padded LDS row stride (f16 elems)
    constexpr int BST = 72;
    __shared__ h16 As[2][128*AST];            // [buf][co_row][k]  128 x 64
    __shared__ h16 Bs[64*BST];                // [pos][k]           64 x 64

    const int co0 = blockIdx.x * 128;
    const int h   = blockIdx.y;
    const int n   = blockIdx.z;
    const int t   = threadIdx.x;
    const int lane = t & 31;
    const int wv   = t >> 5;
    const int wm   = wv >> 1;                 // 0..3 : 32-row co sub-tile
    const int wnn  = wv & 1;                  // 0..1 : 32-col pos sub-tile
    const int l15  = lane & 15;
    const int half = lane >> 4;

    v8f acc00 = {0,0,0,0,0,0,0,0};
    v8f acc01 = acc00, acc10 = acc00, acc11 = acc00;

    // A-tile: thread -> (row = t>>1, 64B half-row by t&1); 4 async b128 per stage
    const int arow = t >> 1;                  // 0..127
    const int ach  = (t & 1) * 32;            // f16 elems
    const h16* asrc = wagg + ((size_t)(n*CO_ + co0 + arow))*KTOT_ + ach;
    h16* adst0 = &As[0][arow*AST + ach];
    h16* adst1 = &As[1][arow*AST + ach];

    // B-tile: thread -> (one position p = t>>2, 16 consecutive k = (t&3)*16)
    const int bp  = t >> 2;                   // 0..63
    const int bq  = (t & 3) * 16;             // k sub-chunk base (= ci offset)
    h16* bdst = &Bs[bp*BST + bq];

    float bx[16];

    // ---- prologue: stage 0 (tap 0, ci0 0)
    {
        h16* al = adst0;
        ASYNC_CP16(asrc, al, 0);  ASYNC_CP16(asrc, al, 16);
        ASYNC_CP16(asrc, al, 32); ASYNC_CP16(asrc, al, 48);
        const int ih = h - 1, iw = bp - 1;
        const bool valid = ((unsigned)ih < (unsigned)H_) && ((unsigned)iw < (unsigned)W_);
        const float* xp = x + (((size_t)(n*CI_ + bq))*H_ + ih)*W_ + iw;
        #pragma unroll
        for (int j = 0; j < 16; ++j) bx[j] = valid ? xp[(size_t)j*(H_*W_)] : 0.f;
    }

    int buf = 0;
    int kk0 = 0;
    #pragma unroll 1
    for (int tap = 0; tap < 9; ++tap) {
        #pragma unroll 1
        for (int ci0 = 0; ci0 < CI_; ci0 += 64, kk0 += 64) {
            // ---- stage B registers -> LDS; A arrives via ASYNCcnt
            {
                union { h16 e[16]; v4u q[4]; } pk;
                #pragma unroll
                for (int j = 0; j < 16; ++j) pk.e[j] = (h16)bx[j];
                reinterpret_cast<v4u*>(bdst)[0] = pk.q[0];
                reinterpret_cast<v4u*>(bdst)[1] = pk.q[1];
            }
            WAIT_ASYNC();
            __syncthreads();

            // ---- issue next stage (overlaps WMMA): no div/mod, loop-derived decode
            const int kkn = kk0 + 64;
            if (kkn < KTOT_) {
                h16* al = (buf == 0) ? adst1 : adst0;
                const h16* ag = asrc + kkn;
                ASYNC_CP16(ag, al, 0);  ASYNC_CP16(ag, al, 16);
                ASYNC_CP16(ag, al, 32); ASYNC_CP16(ag, al, 48);
                const bool same = (ci0 + 64 < CI_);
                const int tapn = same ? tap : tap + 1;
                const int cin  = same ? ci0 + 64 : 0;
                const int khn  = tap_kh(tapn);
                const int kwn  = tapn - 3*khn;
                const int ih   = h + khn - 1;
                const int iw   = bp + kwn - 1;
                const bool valid = ((unsigned)ih < (unsigned)H_) &&
                                   ((unsigned)iw < (unsigned)W_);
                const float* xp = x + (((size_t)(n*CI_ + cin + bq))*H_ + ih)*W_ + iw;
                #pragma unroll
                for (int j = 0; j < 16; ++j) bx[j] = valid ? xp[(size_t)j*(H_*W_)] : 0.f;
            }

            // ---- 8 WMMA over two 32-K slices
            const h16* abuf = As[buf];
            #pragma unroll
            for (int ks = 0; ks < 2; ++ks) {
                union { v16h v; v4u q[2]; } fa0, fa1, fb0, fb1;
                const h16* ar0 = abuf + (wm*32      + l15)*AST + ks*32;
                const h16* ar1 = abuf + (wm*32 + 16 + l15)*AST + ks*32;
                fa0.q[0] = *reinterpret_cast<const v4u*>(ar0 + half*8);
                fa0.q[1] = *reinterpret_cast<const v4u*>(ar0 + 16 + half*8);
                fa1.q[0] = *reinterpret_cast<const v4u*>(ar1 + half*8);
                fa1.q[1] = *reinterpret_cast<const v4u*>(ar1 + 16 + half*8);
                const h16* br0 = &Bs[(wnn*32      + l15)*BST + ks*32 + half*8];
                const h16* br1 = &Bs[(wnn*32 + 16 + l15)*BST + ks*32 + half*8];
                fb0.q[0] = *reinterpret_cast<const v4u*>(br0);
                fb0.q[1] = *reinterpret_cast<const v4u*>(br0 + 16);
                fb1.q[0] = *reinterpret_cast<const v4u*>(br1);
                fb1.q[1] = *reinterpret_cast<const v4u*>(br1 + 16);
                acc00 = __builtin_amdgcn_wmma_f32_16x16x32_f16(false, fa0.v, false, fb0.v,
                                                               (short)0, acc00, false, false);
                acc01 = __builtin_amdgcn_wmma_f32_16x16x32_f16(false, fa0.v, false, fb1.v,
                                                               (short)0, acc01, false, false);
                acc10 = __builtin_amdgcn_wmma_f32_16x16x32_f16(false, fa1.v, false, fb0.v,
                                                               (short)0, acc10, false, false);
                acc11 = __builtin_amdgcn_wmma_f32_16x16x32_f16(false, fa1.v, false, fb1.v,
                                                               (short)0, acc11, false, false);
            }
            __syncthreads();
            buf ^= 1;
        }
    }

    // ---- epilogue: C/D layout (VGPR r -> M=r / M=8+r by lane half)
    const int pos0 = wnn*32 + l15;
    #pragma unroll
    for (int r = 0; r < 8; ++r) {
        {
            const int co = co0 + wm*32 + half*8 + r;
            const float bias = bn[n*CO_ + co];
            const size_t base = (((size_t)(n*CO_ + co))*H_ + h)*W_;
            y[base + pos0]      = acc00[r] + bias;
            y[base + pos0 + 16] = acc01[r] + bias;
        }
        {
            const int co = co0 + wm*32 + 16 + half*8 + r;
            const float bias = bn[n*CO_ + co];
            const size_t base = (((size_t)(n*CO_ + co))*H_ + h)*W_;
            y[base + pos0]      = acc10[r] + bias;
            y[base + pos0 + 16] = acc11[r] + bias;
        }
    }
}

extern "C" void kernel_launch(void* const* d_in, const int* in_sizes, int n_in,
                              void* d_out, int out_size, void* d_ws, size_t ws_size,
                              hipStream_t stream) {
    (void)in_sizes; (void)n_in; (void)out_size; (void)ws_size;
    const float* x     = (const float*)d_in[0];
    const float* Wbank = (const float*)d_in[1];
    const float* Bbank = (const float*)d_in[2];
    const float* w1    = (const float*)d_in[3];
    const float* b1    = (const float*)d_in[4];
    const float* w2    = (const float*)d_in[5];
    const float* b2    = (const float*)d_in[6];
    float* y = (float*)d_out;

    float* pooled = (float*)d_ws;                 // N*CI
    float* pi     = pooled + N_*CI_;              // N*M
    float* bn     = pi + N_*M_;                   // N*CO
    h16*   wagg   = (h16*)(bn + N_*CO_);          // N*CO*KTOT (16B-aligned)

    k_pool<<<N_*CI_, 256, 0, stream>>>(x, pooled);
    k_attn<<<N_, HID_, 0, stream>>>(pooled, w1, b1, w2, b2, Bbank, pi, bn);
    k_aggw<<<N_*CO_, 256, 0, stream>>>(Wbank, pi, wagg);
    dim3 grid(CO_/128, H_, N_);
    k_conv<<<grid, 256, 0, stream>>>(x, wagg, bn, y);
}
